// MILossGaussian_78503412236568
// MI455X (gfx1250) — matrix operand order repaired and checked by reading, not compiled, not run on hardware
//
#include <hip/hip_runtime.h>

typedef __attribute__((ext_vector_type(16))) _Float16 v16h;
typedef __attribute__((ext_vector_type(8)))  float    v8f;

#define P_TOTAL 884736          // 96*96*96 (channel dim folded into P)
#define NBATCH  2
#define NBINS   64
#define CHUNKS  (P_TOTAL / 32)  // 27648 K-chunks of 32 samples
#define GRIDX   96              // 96 blocks * 8 waves = 768 waves -> 36 chunks/wave exactly
#define WSCALE  2.64711e-3f     // sigma / sqrt(2*pi)
#define EXPK    16384.0f        // 1/(2*sigma^2*ln2) == 4/binwidth^2 (exact)
#define INV63   (1.0f / 63.0f)  // linspace(0,1,64) step
#define EPSF    1e-10f
#define LN2F    0.69314718056f

// ---------------------------------------------------------------------------
// Zero the 2 x 64 x 64 f32 histogram in workspace (d_ws is poisoned 0xAA).
// ---------------------------------------------------------------------------
__global__ __launch_bounds__(256) void mi_zero_ws(float* __restrict__ h) {
  int i = blockIdx.x * 256 + threadIdx.x;
  if (i < NBATCH * NBINS * NBINS) h[i] = 0.0f;
}

// ---------------------------------------------------------------------------
// Joint histogram via WMMA:  hist[n] += Wx(16x32 f16) * Wy(32x16 f16)
// Window values are generated on the fly, in fragment layout, one eval each.
//
// f16 A 16x32 layout (per lane, v16h slot j -> K index):
//   lanes 0-15 : row M=lane,    K = {0..15} (contiguous order used here)
//   lanes 16-31: row M=lane-16, K = {16..31}
// f16 B 32x16 layout:
//   lanes 0-15 : col N=lane,    K = {0..15} (this code's contiguous order, same as A)
//   lanes 16-31: col N=lane-16, K = {16..31}
// f32 C 16x16 layout: VGPR r -> row r (lanes 0-15) / r+8 (lanes 16-31), col = lane%16
// ---------------------------------------------------------------------------
__global__ __launch_bounds__(256) void mi_joint_hist(const float* __restrict__ x,
                                                     const float* __restrict__ y,
                                                     float* __restrict__ ghist) {
  __shared__ float lhist[NBINS * NBINS];

  const int n    = blockIdx.y;
  const int lane = threadIdx.x & 31;
  const int wave = threadIdx.x >> 5;
  const int hiA  = (lane & 16) ? 8 : 0;    // unused: A and B fragments both take k base hiB
  const int hiB  = (lane & 16) ? 16 : 0;   // B-fragment K base
  const int bcol = lane & 15;              // bin index within a 16-wide tile

  const float* __restrict__ xb = x + (size_t)n * P_TOTAL;
  const float* __restrict__ yb = y + (size_t)n * P_TOTAL;

  // 4x4 tiles of 16x16 f32 accumulators = full 64x64 joint histogram partial
  v8f c[16];
#pragma unroll
  for (int t = 0; t < 16; ++t)
#pragma unroll
    for (int r = 0; r < 8; ++r) c[t][r] = 0.0f;

  for (int chunk = blockIdx.x * 8 + wave; chunk < CHUNKS; chunk += GRIDX * 8) {
    const int p0 = chunk * 32;
    const float xv = xb[p0 + lane];   // coalesced 128B per wave
    const float yv = yb[p0 + lane];

    // Redistribute the 32 samples into fragment k-order via lane permutes.
    float xk[16], yk[16];
#pragma unroll
    for (int j = 0; j < 16; ++j) {
      const int kA = hiB + j;
      xk[j] = __shfl(xv, kA, 32);
      yk[j] = __shfl(yv, hiB + j, 32);
    }

    // Build all 4 B tiles (y-window, 4 bin-column groups).
    v16h b[4];
#pragma unroll
    for (int tj = 0; tj < 4; ++tj) {
      const float bin = (float)(tj * 16 + bcol) * INV63;
#pragma unroll
      for (int j = 0; j < 16; ++j) {
        const float d = yk[j] - bin;
        const float w = WSCALE * __builtin_amdgcn_exp2f(-(d * d) * EXPK);
        b[tj][j] = (_Float16)w;
      }
    }

    // For each A tile (x-window, bin-row group): build fragment, 4 WMMAs.
#pragma unroll
    for (int ti = 0; ti < 4; ++ti) {
      const float bin = (float)(ti * 16 + bcol) * INV63;
      v16h a;
#pragma unroll
      for (int j = 0; j < 16; ++j) {
        const float d = xk[j] - bin;
        const float w = WSCALE * __builtin_amdgcn_exp2f(-(d * d) * EXPK);
        a[j] = (_Float16)w;
      }
#pragma unroll
      for (int tj = 0; tj < 4; ++tj) {
        c[ti * 4 + tj] = __builtin_amdgcn_wmma_f32_16x16x32_f16(
            /*neg_a=*/false, a, /*neg_b=*/false, b[tj],
            /*c_mod=*/(short)0, c[ti * 4 + tj],
            /*reuse_a=*/false, /*reuse_b=*/false);
      }
    }
  }

  // Reduce the 8 waves' accumulators through LDS, then one global push.
  for (int i = threadIdx.x; i < NBINS * NBINS; i += 256) lhist[i] = 0.0f;
  __syncthreads();

  const int rOff = (lane & 16) ? 8 : 0;
#pragma unroll
  for (int ti = 0; ti < 4; ++ti)
#pragma unroll
    for (int tj = 0; tj < 4; ++tj)
#pragma unroll
      for (int r = 0; r < 8; ++r) {
        const int row = ti * 16 + r + rOff;
        const int col = tj * 16 + bcol;
        atomicAdd(&lhist[row * NBINS + col], c[ti * 4 + tj][r]);
      }
  __syncthreads();

  float* __restrict__ gh = ghist + (size_t)n * NBINS * NBINS;
  for (int i = threadIdx.x; i < NBINS * NBINS; i += 256)
    atomicAdd(&gh[i], lhist[i]);
}

// ---------------------------------------------------------------------------
// Entropy finalize: tiny 64x64 reductions -> scalar NMI loss.
// ---------------------------------------------------------------------------
__device__ __forceinline__ float block_reduce_sum(float v, float* buf) {
  const int t = threadIdx.x;
  buf[t] = v;
  __syncthreads();
  for (int s = 128; s > 0; s >>= 1) {
    if (t < s) buf[t] += buf[t + s];
    __syncthreads();
  }
  const float r = buf[0];
  __syncthreads();
  return r;
}

__global__ __launch_bounds__(256) void mi_finalize(const float* __restrict__ ghist,
                                                   float* __restrict__ out) {
  __shared__ float buf[256];
  __shared__ float res[NBATCH];
  const int t = threadIdx.x;

  for (int n = 0; n < NBATCH; ++n) {
    const float* __restrict__ h = ghist + (size_t)n * NBINS * NBINS;

    float s = 0.0f;
    for (int i = t; i < NBINS * NBINS; i += 256) s += h[i];
    const float total = block_reduce_sum(s, buf) + EPSF;
    const float inv = 1.0f / total;

    // joint entropy
    float ejp = 0.0f;
    for (int i = t; i < NBINS * NBINS; i += 256) {
      const float p = h[i] * inv;
      ejp += p * (__builtin_amdgcn_logf(p + EPSF) * LN2F);
    }
    const float ent_j = -block_reduce_sum(ejp, buf);

    // marginal entropies (thread t < 64 owns row/col t)
    float exp_ = 0.0f, eyp = 0.0f;
    if (t < NBINS) {
      float rs = 0.0f, cs = 0.0f;
      for (int j = 0; j < NBINS; ++j) {
        rs += h[t * NBINS + j];
        cs += h[j * NBINS + t];
      }
      rs *= inv;
      cs *= inv;
      exp_ = rs * (__builtin_amdgcn_logf(rs + EPSF) * LN2F);
      eyp  = cs * (__builtin_amdgcn_logf(cs + EPSF) * LN2F);
    }
    const float ent_x = -block_reduce_sum(exp_, buf);
    const float ent_y = -block_reduce_sum(eyp, buf);

    if (t == 0) res[n] = (ent_x + ent_y) / ent_j;
    __syncthreads();
  }

  if (t == 0) out[0] = -0.5f * (res[0] + res[1]);
}

// ---------------------------------------------------------------------------
extern "C" void kernel_launch(void* const* d_in, const int* in_sizes, int n_in,
                              void* d_out, int out_size, void* d_ws, size_t ws_size,
                              hipStream_t stream) {
  const float* x = (const float*)d_in[0];
  const float* y = (const float*)d_in[1];
  float* out   = (float*)d_out;
  float* ghist = (float*)d_ws;   // 2 * 64 * 64 f32 = 32 KB

  mi_zero_ws<<<dim3(32), 256, 0, stream>>>(ghist);
  mi_joint_hist<<<dim3(GRIDX, NBATCH), 256, 0, stream>>>(x, y, ghist);
  mi_finalize<<<dim3(1), 256, 0, stream>>>(ghist, out);
}
